// CE_Block_Shared_39092792328573
// MI455X (gfx1250) — matrix operand order repaired and checked by reading, not compiled, not run on hardware
//
#include <hip/hip_runtime.h>
#include <hip/hip_bf16.h>
#include <math.h>

typedef __bf16 bf16;
typedef __attribute__((ext_vector_type(16))) bf16  v16bf;
typedef __attribute__((ext_vector_type(8)))  float v8f;

#define HN       12
#define CDIM     768
#define NTOK     384
#define LMTOK    128
#define LSTOK    256
#define BATCH    64
#define HDIM     64
#define LKEEP    180
#define LREM     76
#define NKEEP    308     /* 128 + 180 */
#define HIDDIM   3072
#define QKVC     2304

#define BM   128
#define BN   128
#define BKK  32
#define LDSK 40          /* padded LDS row stride in bf16 */

__device__ __forceinline__ bf16 f2bf(float f) {
    unsigned u = __float_as_uint(f);
    unsigned r = u + 0x7fffu + ((u >> 16) & 1u);
    union { unsigned short s; bf16 b; } cv;
    cv.s = (unsigned short)(r >> 16);
    return cv.b;
}

/* ---- gfx1250 async global->LDS copy (ASYNCcnt-tracked, bypasses VGPRs) ---- */
__device__ __forceinline__ void async_b128(void* lds, const void* gptr) {
    unsigned loff = (unsigned)(unsigned long long)lds;   /* LDS byte offset = addr[31:0] */
    asm volatile("global_load_async_to_lds_b128 %0, %1, off"
                 :: "v"(loff), "v"(gptr)
                 : "memory");
}
__device__ __forceinline__ void wait_async(int n) {      /* n is workgroup-uniform */
    if (n >= 4)      asm volatile("s_wait_asynccnt 4" ::: "memory");
    else if (n >= 2) asm volatile("s_wait_asynccnt 2" ::: "memory");
    else             asm volatile("s_wait_asynccnt 0" ::: "memory");
}

/* ---- WMMA fragment loads from LDS (layouts per CDNA5 ISA 7.12.2) ---- */
__device__ __forceinline__ v16bf frag_a(const bf16* s, int mb, int lane) {
    int m  = mb + (lane & 15);
    int kb = (lane >> 4) << 3;            /* 0 or 8 */
    const bf16* p = s + m * LDSK + kb;
    union { v16bf v; uint4 q[2]; } u;
    u.q[0] = *(const uint4*)p;            /* K kb..kb+7   */
    u.q[1] = *(const uint4*)(p + 16);     /* K kb+16..+23 */
    return u.v;
}
__device__ __forceinline__ v16bf frag_b(const bf16* s, int nb, int lane) {
    int n  = nb + (lane & 15);
    int kb = (lane >> 4) << 3;            /* 0 or 8 */
    const bf16* p = s + n * LDSK + kb;
    union { v16bf v; uint4 q[2]; } u;
    u.q[0] = *(const uint4*)p;            /* K kb..kb+7   */
    u.q[1] = *(const uint4*)(p + 16);     /* K kb+16..+23 */
    return u.v;
}

/* ------------------------------------------------------------------ */
/* Generic batched-strided bf16 WMMA GEMM, double-buffered LDS with   */
/* async global->LDS staging overlapped with v_wmma compute.          */
/*   D[M,Nn] = alpha * A[M,K] x B(T)[Nn,K] (+bias)(GELU)(+Res)        */
/* batch index z -> (b = z/Hdiv, h = z%Hdiv); offsets b*s?b + h*s?h   */
/* ------------------------------------------------------------------ */
__global__ __launch_bounds__(256, 2)
void gemm_bf16_kernel(const bf16* __restrict__ A, long long sAb, long long sAh, int lda,
                      const bf16* __restrict__ Bm, long long sBb, long long sBh, int ldb, int bT,
                      const float* __restrict__ bias,
                      const float* __restrict__ Res, long long sRb, long long sRh, int ldr,
                      float* __restrict__ Of, bf16* __restrict__ Ob,
                      long long sCb, long long sCh, int ldc,
                      int M, int Nn, int K, float alpha, int Hdiv, int gelu)
{
    __shared__ __align__(16) bf16 As[2][BM * LDSK];
    __shared__ __align__(16) bf16 Bs[2][BN * LDSK];

    const int tid  = threadIdx.x;
    const int lane = tid & 31;
    const int wid  = tid >> 5;
    const int wm   = wid & 1;      /* 2 wave-rows  x 64 rows  */
    const int wn   = wid >> 1;     /* 4 wave-cols  x 32 cols  */

    const int z  = blockIdx.z;
    const int bb = z / Hdiv, hh = z % Hdiv;
    const long long offA = (long long)bb * sAb + (long long)hh * sAh;
    const long long offB = (long long)bb * sBb + (long long)hh * sBh;
    const long long offC = (long long)bb * sCb + (long long)hh * sCh;
    const long long offR = (long long)bb * sRb + (long long)hh * sRh;

    const int m0 = blockIdx.y * BM;
    const int n0 = blockIdx.x * BN;

    const bool aFull = (m0 + BM <= M);            /* uniform per block */
    const bool bFull = (!bT) && (n0 + BN <= Nn);  /* uniform per block */

    v8f acc[4][2];
#pragma unroll
    for (int i = 0; i < 4; ++i)
#pragma unroll
        for (int j = 0; j < 2; ++j)
#pragma unroll
            for (int e = 0; e < 8; ++e) acc[i][j][e] = 0.0f;

    /* stage tile for k0 into buffer buf; returns #async instructions issued per wave */
    auto stage = [&](int buf, int k0) -> int {
        int issued = 0;
        /* ---- A tile [BM x BKK] ---- */
#pragma unroll
        for (int it = 0; it < 2; ++it) {
            int c  = tid + it * 256;          /* 512 chunks of 8 bf16 */
            int r  = c >> 2;
            int kc = (c & 3) << 3;
            int gm = m0 + r;
            const bf16* gp = A + offA + (long long)gm * lda + (k0 + kc);
            bf16* lp = &As[buf][r * LDSK + kc];
            if (aFull) { async_b128(lp, gp); }
            else {
                uint4 val = {0u, 0u, 0u, 0u};
                if (gm < M) val = *(const uint4*)gp;
                *(uint4*)lp = val;
            }
        }
        if (aFull) issued += 2;
        /* ---- B tile [BN x BKK] ---- */
        if (!bT) {
#pragma unroll
            for (int it = 0; it < 2; ++it) {
                int c  = tid + it * 256;
                int r  = c >> 2;
                int kc = (c & 3) << 3;
                int gn = n0 + r;
                const bf16* gp = Bm + offB + (long long)gn * ldb + (k0 + kc);
                bf16* lp = &Bs[buf][r * LDSK + kc];
                if (bFull) { async_b128(lp, gp); }
                else {
                    uint4 val = {0u, 0u, 0u, 0u};
                    if (gn < Nn) val = *(const uint4*)gp;
                    *(uint4*)lp = val;
                }
            }
            if (bFull) issued += 2;
        } else {
            /* B stored [K][Nn]: transpose into LDS (guarded, synchronous) */
#pragma unroll
            for (int it = 0; it < 2; ++it) {
                int c   = tid + it * 256;     /* 32 k-rows x 16 n-chunks */
                int kk  = c >> 4;
                int nc  = (c & 15) << 3;
                int gn0 = n0 + nc;
                uint4 val = {0u, 0u, 0u, 0u};
                if (gn0 < Nn)
                    val = *(const uint4*)(Bm + offB + (long long)(k0 + kk) * ldb + gn0);
                const bf16* t = (const bf16*)&val;
#pragma unroll
                for (int i = 0; i < 8; ++i) Bs[buf][(nc + i) * LDSK + kk] = t[i];
            }
        }
        return issued;
    };

    const int nIter = K / BKK;       /* all K in this workload are multiples of 32 */
    stage(0, 0);
    for (int i = 0; i < nIter; ++i) {
        const int cur = i & 1;
        int issuedNext = 0;
        if (i + 1 < nIter)
            issuedNext = stage(cur ^ 1, (i + 1) * BKK);
        /* all ops older than the newly issued ones (i.e. current tile) complete */
        wait_async(issuedNext);
        __syncthreads();

        v16bf af[4], bfv[2];
#pragma unroll
        for (int mi = 0; mi < 4; ++mi) af[mi]  = frag_a(As[cur], wm * 64 + mi * 16, lane);
#pragma unroll
        for (int ni = 0; ni < 2; ++ni) bfv[ni] = frag_b(Bs[cur], wn * 32 + ni * 16, lane);
#pragma unroll
        for (int mi = 0; mi < 4; ++mi)
#pragma unroll
            for (int ni = 0; ni < 2; ++ni)
                acc[mi][ni] = __builtin_amdgcn_wmma_f32_16x16x32_bf16(
                    false, af[mi], false, bfv[ni], (short)0, acc[mi][ni], false, false);
        __syncthreads();   /* protect buf 'cur' before it is re-staged next iter */
    }

    /* ---- epilogue: C/D layout = lane(0..15)->N, halves->M+8, vgpr->M ---- */
    const int rhalf = (lane >> 4) << 3;
    const int colL  = lane & 15;
#pragma unroll
    for (int mi = 0; mi < 4; ++mi) {
#pragma unroll
        for (int ni = 0; ni < 2; ++ni) {
            int nloc = wn * 32 + ni * 16 + colL;
            int gn   = n0 + nloc;
            if (gn >= Nn) continue;
            float bv = bias ? bias[gn] : 0.0f;
#pragma unroll
            for (int r = 0; r < 8; ++r) {
                int gm = m0 + wm * 64 + mi * 16 + rhalf + r;
                if (gm >= M) continue;
                float v = acc[mi][ni][r] * alpha + bv;
                if (gelu) v = 0.5f * v * (1.0f + erff(v * 0.70710678118f));
                if (Res) v += Res[offR + (long long)gm * ldr + gn];
                long long o = offC + (long long)gm * ldc + gn;
                if (Of) Of[o] = v;
                if (Ob) Ob[o] = f2bf(v);
            }
        }
    }
}

/* ---------------- LayerNorm (C=768) -> bf16 ---------------- */
__global__ __launch_bounds__(256)
void ln_kernel(const float* __restrict__ x, const float* __restrict__ g,
               const float* __restrict__ be, bf16* __restrict__ out)
{
    const int row = blockIdx.x, tid = threadIdx.x;
    const float* xr = x + (long long)row * CDIM;
    __shared__ float red[256];
    float v0 = xr[tid], v1 = xr[tid + 256], v2 = xr[tid + 512];
    red[tid] = v0 + v1 + v2;
    __syncthreads();
    for (int o = 128; o; o >>= 1) { if (tid < o) red[tid] += red[tid + o]; __syncthreads(); }
    float mean = red[0] * (1.0f / 768.0f);
    __syncthreads();
    float d0 = v0 - mean, d1 = v1 - mean, d2 = v2 - mean;
    red[tid] = d0 * d0 + d1 * d1 + d2 * d2;
    __syncthreads();
    for (int o = 128; o; o >>= 1) { if (tid < o) red[tid] += red[tid + o]; __syncthreads(); }
    float rstd = rsqrtf(red[0] * (1.0f / 768.0f) + 1e-5f);
    bf16* orow = out + (long long)row * CDIM;
    orow[tid]       = f2bf(d0 * rstd * g[tid]       + be[tid]);
    orow[tid + 256] = f2bf(d1 * rstd * g[tid + 256] + be[tid + 256]);
    orow[tid + 512] = f2bf(d2 * rstd * g[tid + 512] + be[tid + 512]);
}

/* ---------------- f32 -> bf16 convert ---------------- */
__global__ void cvt_kernel(const float* __restrict__ src, bf16* __restrict__ dst, long long n)
{
    long long i = (long long)blockIdx.x * 256 + threadIdx.x;
    if (i < n) dst[i] = f2bf(src[i]);
}

/* ---------------- row softmax (in-place f32, optional bf16 probs) ---------------- */
__global__ __launch_bounds__(128)
void softmax_kernel(float* __restrict__ S, bf16* __restrict__ P, int width)
{
    long long row = blockIdx.x;
    float* r = S + row * width;
    const int tid = threadIdx.x;
    __shared__ float red[128];
    float m = -1e30f;
    for (int i = tid; i < width; i += 128) m = fmaxf(m, r[i]);
    red[tid] = m; __syncthreads();
    for (int o = 64; o; o >>= 1) { if (tid < o) red[tid] = fmaxf(red[tid], red[tid + o]); __syncthreads(); }
    m = red[0]; __syncthreads();
    float s = 0.0f;
    for (int i = tid; i < width; i += 128) { float e = __expf(r[i] - m); r[i] = e; s += e; }
    red[tid] = s; __syncthreads();
    for (int o = 64; o; o >>= 1) { if (tid < o) red[tid] += red[tid + o]; __syncthreads(); }
    float inv = 1.0f / red[0];
    for (int i = tid; i < width; i += 128) {
        float p = r[i] * inv;
        r[i] = p;
        if (P) P[row * width + i] = f2bf(p);
    }
}

/* ---------------- a = mean over (H, q) of attn_t2s ---------------- */
__global__ __launch_bounds__(128)
void colmean_kernel(const float* __restrict__ attn, float* __restrict__ a)
{
    int b = blockIdx.y;
    int c = blockIdx.x * 128 + threadIdx.x;          /* 0..511 */
    const float* base = attn + (long long)b * HN * 256 * 512 + c;
    float s = 0.0f;
    for (int h = 0; h < HN; ++h)
        for (int q = 0; q < 256; ++q)
            s += base[((long long)h * 256 + q) * 512];
    a[b * 512 + c] = s * (1.0f / (HN * 256.0f));
}

/* ---------------- stable descending rank sort (matches argsort(-a)) ---------------- */
__global__ __launch_bounds__(LSTOK)
void topk_kernel(const float* __restrict__ a, int coloff,
                 int* __restrict__ keep, int* __restrict__ rem)
{
    int b = blockIdx.x, i = threadIdx.x;
    __shared__ float av[LSTOK];
    __shared__ int ord[LSTOK];
    av[i] = a[b * 512 + coloff + i];
    __syncthreads();
    float vi = av[i];
    int rank = 0;
    for (int j = 0; j < LSTOK; ++j) {
        float vj = av[j];
        rank += (vj > vi) || (vj == vi && j < i);
    }
    ord[rank] = i;
    __syncthreads();
    if (i < LKEEP) keep[b * LKEEP + i] = ord[i];
    else           rem[b * LREM + (i - LKEEP)] = ord[i];
}

/* ---------------- token gather after top-k ---------------- */
__global__ __launch_bounds__(256)
void gather_kernel(const float* __restrict__ xres, const int* __restrict__ keepV,
                   const int* __restrict__ keepI, float* __restrict__ xg)
{
    int l = blockIdx.x, b = blockIdx.y, modal = blockIdx.z;
    const int* keep = modal ? keepI : keepV;
    int tok = (l < LMTOK) ? l : (LMTOK + keep[b * LKEEP + (l - LMTOK)]);
    const float* src = xres + (((long long)modal * BATCH + b) * NTOK + tok) * CDIM;
    float* dst = xg + (((long long)modal * BATCH + b) * NKEEP + l) * CDIM;
    for (int c = threadIdx.x; c < CDIM; c += 256) dst[c] = src[c];
}

/* ---------------- pack per-(b,h) head slices into contiguous [B*H, totrows, 64] ---------------- */
__global__ __launch_bounds__(256)
void pack_kernel(bf16* __restrict__ dst, const bf16* __restrict__ Y, long long ybase,
                 int coloff, int tok0, int nrows, int rowoff, int totrows)
{
    long long idx = (long long)blockIdx.x * 256 + threadIdx.x;
    long long total = (long long)BATCH * HN * nrows * HDIM;
    if (idx >= total) return;
    int d = idx & 63;
    long long t = idx >> 6;
    int r = (int)(t % nrows); t /= nrows;
    int h = (int)(t % HN);
    int b = (int)(t / HN);
    dst[(((long long)(b * HN + h)) * totrows + rowoff + r) * HDIM + d] =
        Y[ybase + ((long long)b * NTOK + tok0 + r) * QKVC + coloff + h * HDIM + d];
}

/* ================================================================== */
extern "C" void kernel_launch(void* const* d_in, const int* in_sizes, int n_in,
                              void* d_out, int out_size, void* d_ws, size_t ws_size,
                              hipStream_t stream)
{
    const float* x_v   = (const float*)d_in[0];
    const float* x_i   = (const float*)d_in[1];
    const float* g1v   = (const float*)d_in[8];
    const float* b1v   = (const float*)d_in[9];
    const float* g1i   = (const float*)d_in[10];
    const float* b1i   = (const float*)d_in[11];
    const float* Wqkv  = (const float*)d_in[12];
    const float* Wproj = (const float*)d_in[13];
    const float* bproj = (const float*)d_in[14];
    const float* g2v   = (const float*)d_in[15];
    const float* b2v   = (const float*)d_in[16];
    const float* g2i   = (const float*)d_in[17];
    const float* b2i   = (const float*)d_in[18];
    const float* Wfc1  = (const float*)d_in[19];
    const float* bfc1  = (const float*)d_in[20];
    const float* Wfc2  = (const float*)d_in[21];
    const float* bfc2  = (const float*)d_in[22];

    /* ---- d_out layout (flat, return order) ---- */
    const long long XOUT  = (long long)BATCH * NKEEP * CDIM;       /* 15,138,816 */
    float* out_x    = (float*)d_out;                               /* x_v then x_i */
    int*   keep_v   = (int*)((float*)d_out + 2 * XOUT);
    int*   keep_i   = keep_v + BATCH * LKEEP;
    int*   rem_v    = keep_i + BATCH * LKEEP;
    int*   rem_i    = rem_v + BATCH * LREM;
    float* attn_out = (float*)(rem_i + BATCH * LREM);              /* [B,H,256,512] */

    /* ---- workspace bump allocator ---- */
    char* ws = (char*)d_ws;
    size_t off = 0;
    auto alloc = [&](size_t bytes) -> void* {
        void* p = ws + off;
        off = (off + bytes + 255) & ~(size_t)255;
        return p;
    };
    const long long XN_E   = 2LL * BATCH * NTOK * CDIM;        /* 37,748,736 */
    const long long Y_E    = 2LL * BATCH * NTOK * QKVC;        /* 113,246,208 */
    bf16*  xn     = (bf16*)alloc(XN_E * 2);
    bf16*  Ybf    = (bf16*)alloc(Y_E * 2);
    bf16*  Wq_b   = (bf16*)alloc((long long)QKVC * CDIM * 2);
    bf16*  Wp_b   = (bf16*)alloc((long long)CDIM * CDIM * 2);
    bf16*  W1_b   = (bf16*)alloc((long long)HIDDIM * CDIM * 2);
    bf16*  W2_b   = (bf16*)alloc((long long)CDIM * HIDDIM * 2);
    float* Smt    = (float*)alloc((long long)BATCH * HN * 128 * 128 * 4);
    bf16*  Pmt    = (bf16*)alloc((long long)BATCH * HN * 128 * 128 * 2);
    float* Ss     = (float*)alloc((long long)BATCH * HN * 256 * 512 * 4);
    bf16*  Ps     = (bf16*)alloc((long long)BATCH * HN * 256 * 512 * 2);
    bf16*  Kcat   = (bf16*)alloc((long long)BATCH * HN * 512 * HDIM * 2);
    bf16*  Vcat   = (bf16*)alloc((long long)BATCH * HN * 512 * HDIM * 2);
    bf16*  mrg_b  = (bf16*)alloc(XN_E * 2);
    float* xres   = (float*)alloc(XN_E * 4);
    float* avec   = (float*)alloc((long long)BATCH * 512 * 4);
    float* xg     = (float*)alloc(2LL * BATCH * NKEEP * CDIM * 4);
    bf16*  xgn    = (bf16*)alloc(2LL * BATCH * NKEEP * CDIM * 2);
    bf16*  hbuf   = (bf16*)alloc(2LL * BATCH * NKEEP * HIDDIM * 2);

    const float scale = 0.125f;   /* 64^-0.5 */
    const long long sYtok = (long long)NTOK * QKVC;   /* per-batch stride in Ybf */
    const long long modY  = (long long)BATCH * sYtok; /* per-modality stride     */
    const long long modM  = (long long)BATCH * NTOK * CDIM;

    /* 1) weight conversion */
    {
        long long n;
        n = (long long)QKVC * CDIM;   cvt_kernel<<<dim3((unsigned)((n + 255) / 256)), 256, 0, stream>>>(Wqkv, Wq_b, n);
        n = (long long)CDIM * CDIM;   cvt_kernel<<<dim3((unsigned)((n + 255) / 256)), 256, 0, stream>>>(Wproj, Wp_b, n);
        n = (long long)HIDDIM * CDIM; cvt_kernel<<<dim3((unsigned)((n + 255) / 256)), 256, 0, stream>>>(Wfc1, W1_b, n);
        n = (long long)CDIM * HIDDIM; cvt_kernel<<<dim3((unsigned)((n + 255) / 256)), 256, 0, stream>>>(Wfc2, W2_b, n);
    }
    /* 2) LN1 */
    ln_kernel<<<dim3(BATCH * NTOK), 256, 0, stream>>>(x_v, g1v, b1v, xn);
    ln_kernel<<<dim3(BATCH * NTOK), 256, 0, stream>>>(x_i, g1i, b1i, xn + modM);

    /* 3) QKV: [49152,768] x Wqkv[2304,768]^T -> Ybf (bf16) */
    gemm_bf16_kernel<<<dim3(QKVC / BN, (2 * BATCH * NTOK) / BM, 1), 256, 0, stream>>>(
        xn, 0, 0, CDIM,  Wq_b, 0, 0, CDIM, 0,
        nullptr, nullptr, 0, 0, 0,
        nullptr, Ybf, 0, 0, QKVC,
        2 * BATCH * NTOK, QKVC, CDIM, 1.0f, 1, 0);

    /* 4) per-modality attention */
    for (int modal = 0; modal < 2; ++modal) {
        const bf16* Ym = Ybf + modal * modY;
        /* --- mt attention: scores = Qmt x Kmt^T --- */
        gemm_bf16_kernel<<<dim3(1, 1, BATCH * HN), 256, 0, stream>>>(
            Ym, sYtok, 64, QKVC,
            Ym + 768, sYtok, 64, QKVC, 0,
            nullptr, nullptr, 0, 0, 0,
            Smt, nullptr, (long long)HN * 128 * 128, 128 * 128, 128,
            128, 128, HDIM, scale, HN, 0);
        softmax_kernel<<<dim3(BATCH * HN * 128), 128, 0, stream>>>(Smt, Pmt, 128);
        /* out_mt = P x V (V is [K=tok][N=d] -> bT) */
        gemm_bf16_kernel<<<dim3(1, 1, BATCH * HN), 256, 0, stream>>>(
            Pmt, (long long)HN * 128 * 128, 128 * 128, 128,
            Ym + 1536, sYtok, 64, QKVC, 1,
            nullptr, nullptr, 0, 0, 0,
            nullptr, mrg_b + modal * modM, (long long)NTOK * CDIM, 64, CDIM,
            128, HDIM, 128, 1.0f, HN, 0);

        /* --- s attention: KV = concat(k_mt_V, k_mt_I, k_s_own) --- */
        long long np;
        np = (long long)BATCH * HN * 128 * HDIM;
        pack_kernel<<<dim3((unsigned)((np + 255) / 256)), 256, 0, stream>>>(Kcat, Ybf, 0,     768, 0, 128, 0, 512);
        pack_kernel<<<dim3((unsigned)((np + 255) / 256)), 256, 0, stream>>>(Kcat, Ybf, modY,  768, 0, 128, 128, 512);
        pack_kernel<<<dim3((unsigned)((np + 255) / 256)), 256, 0, stream>>>(Vcat, Ybf, 0,    1536, 0, 128, 0, 512);
        pack_kernel<<<dim3((unsigned)((np + 255) / 256)), 256, 0, stream>>>(Vcat, Ybf, modY, 1536, 0, 128, 128, 512);
        np = (long long)BATCH * HN * 256 * HDIM;
        pack_kernel<<<dim3((unsigned)((np + 255) / 256)), 256, 0, stream>>>(Kcat, Ybf, modal * modY,  768, 128, 256, 256, 512);
        pack_kernel<<<dim3((unsigned)((np + 255) / 256)), 256, 0, stream>>>(Vcat, Ybf, modal * modY, 1536, 128, 256, 256, 512);

        gemm_bf16_kernel<<<dim3(512 / BN, 256 / BM, BATCH * HN), 256, 0, stream>>>(
            Ym + 128LL * QKVC, sYtok, 64, QKVC,
            Kcat, (long long)HN * 512 * HDIM, 512 * HDIM, HDIM, 0,
            nullptr, nullptr, 0, 0, 0,
            Ss, nullptr, (long long)HN * 256 * 512, 256 * 512, 512,
            256, 512, HDIM, scale, HN, 0);
        softmax_kernel<<<dim3(BATCH * HN * 256), 128, 0, stream>>>(Ss, Ps, 512);
        gemm_bf16_kernel<<<dim3(1, 256 / BM, BATCH * HN), 256, 0, stream>>>(
            Ps, (long long)HN * 256 * 512, 256 * 512, 512,
            Vcat, (long long)HN * 512 * HDIM, 512 * HDIM, HDIM, 1,
            nullptr, nullptr, 0, 0, 0,
            nullptr, mrg_b + modal * modM + 128LL * CDIM, (long long)NTOK * CDIM, 64, CDIM,
            256, HDIM, 512, 1.0f, HN, 0);
    }

    /* 5) t2s attention map -> d_out attn region (then softmax in place) */
    {
        long long np = (long long)BATCH * HN * 256 * HDIM;
        pack_kernel<<<dim3((unsigned)((np + 255) / 256)), 256, 0, stream>>>(Kcat, Ybf, 0,    768, 128, 256, 0, 512);
        pack_kernel<<<dim3((unsigned)((np + 255) / 256)), 256, 0, stream>>>(Kcat, Ybf, modY, 768, 128, 256, 256, 512);
        for (int modal = 0; modal < 2; ++modal) {
            gemm_bf16_kernel<<<dim3(512 / BN, 1, BATCH * HN), 256, 0, stream>>>(
                Ybf + modal * modY, sYtok, 64, QKVC,
                Kcat, (long long)HN * 512 * HDIM, 512 * HDIM, HDIM, 0,
                nullptr, nullptr, 0, 0, 0,
                attn_out + (long long)modal * 128 * 512, nullptr,
                (long long)HN * 256 * 512, 256 * 512, 512,
                128, 512, HDIM, scale, HN, 0);
        }
        softmax_kernel<<<dim3(BATCH * HN * 256), 128, 0, stream>>>(attn_out, nullptr, 512);
        colmean_kernel<<<dim3(4, BATCH), 128, 0, stream>>>(attn_out, avec);
    }

    /* 6) proj + residual -> xres */
    for (int modal = 0; modal < 2; ++modal) {
        gemm_bf16_kernel<<<dim3(CDIM / BN, (BATCH * NTOK) / BM, 1), 256, 0, stream>>>(
            mrg_b + modal * modM, 0, 0, CDIM,
            Wp_b, 0, 0, CDIM, 0,
            bproj,
            modal ? x_i : x_v, 0, 0, CDIM,
            xres + modal * modM, nullptr, 0, 0, CDIM,
            BATCH * NTOK, CDIM, CDIM, 1.0f, 1, 0);
    }

    /* 7) top-k selection (gidx == arange, so keep == sorted indices) */
    topk_kernel<<<dim3(BATCH), LSTOK, 0, stream>>>(avec, 0,   keep_v, rem_v);
    topk_kernel<<<dim3(BATCH), LSTOK, 0, stream>>>(avec, 256, keep_i, rem_i);

    /* 8) gather kept tokens */
    gather_kernel<<<dim3(NKEEP, BATCH, 2), 256, 0, stream>>>(xres, keep_v, keep_i, xg);

    /* 9) LN2 */
    const long long modG = (long long)BATCH * NKEEP * CDIM;
    ln_kernel<<<dim3(BATCH * NKEEP), 256, 0, stream>>>(xg, g2v, b2v, xgn);
    ln_kernel<<<dim3(BATCH * NKEEP), 256, 0, stream>>>(xg + modG, g2i, b2i, xgn + modG);

    /* 10) MLP: fc1 (+exact GELU) -> fc2 (+residual) -> d_out x region */
    gemm_bf16_kernel<<<dim3(HIDDIM / BN, (2 * BATCH * NKEEP) / BM, 1), 256, 0, stream>>>(
        xgn, 0, 0, CDIM,
        W1_b, 0, 0, CDIM, 0,
        bfc1, nullptr, 0, 0, 0,
        nullptr, hbuf, 0, 0, HIDDIM,
        2 * BATCH * NKEEP, HIDDIM, CDIM, 1.0f, 1, 1);
    gemm_bf16_kernel<<<dim3(CDIM / BN, (2 * BATCH * NKEEP) / BM, 1), 256, 0, stream>>>(
        hbuf, 0, 0, HIDDIM,
        W2_b, 0, 0, HIDDIM, 0,
        bfc2,
        xg, 0, 0, CDIM,
        out_x, nullptr, 0, 0, CDIM,
        2 * BATCH * NKEEP, CDIM, HIDDIM, 1.0f, 1, 0);

    (void)in_sizes; (void)n_in; (void)out_size; (void)ws_size;
}
